// CausalSelfAttention_20942260535391
// MI455X (gfx1250) — matrix operand order repaired and checked by reading, not compiled, not run on hardware
//
#include <hip/hip_runtime.h>

// ---------------------------------------------------------------------------
// CDNA5 (gfx1250) causal self-attention: bf16 WMMA everywhere, f32 accumulate.
// Round 3: 32x64-per-wave GEMM tiles (8 WMMA/k-step, B-fragment reuse),
// 32-row flash-attention Q tiles (full-wave softmax, 16 WMMA per KV chunk),
// async global->LDS staging for the GEMM A tile (ASYNCcnt path).
// ---------------------------------------------------------------------------

typedef __attribute__((ext_vector_type(16))) __bf16 v16bf;
typedef __attribute__((ext_vector_type(8)))  float  v8f;

union Frag { v16bf v; unsigned u[8]; };

constexpr int BATCH = 4;
constexpr int SEQ   = 2048;
constexpr int CH    = 1024;
constexpr int HEADS = 16;
constexpr int HD    = 64;
constexpr int MROWS = BATCH * SEQ;   // 8192
constexpr int NQKV  = 3 * CH;        // 3072

#ifndef USE_ASYNC_LDS_COPY
#define USE_ASYNC_LDS_COPY 1        // global_load_async_to_lds_b128 for A tile
#endif

// low 32 bits of a generic shared-aperture pointer == LDS byte offset
__device__ __forceinline__ unsigned lds_offset(const void* p) {
  return (unsigned)(unsigned long long)p;
}

__device__ __forceinline__ void async_wait0() {
#if __has_builtin(__builtin_amdgcn_s_wait_asynccnt)
  __builtin_amdgcn_s_wait_asynccnt(0);
#else
  asm volatile("s_wait_asynccnt 0x0" ::: "memory");
#endif
}

__device__ __forceinline__ void pack8(const uint4& lo, const uint4& hi,
                                      unsigned* pk) {
  const unsigned short* a = (const unsigned short*)&lo;
  const unsigned short* b = (const unsigned short*)&hi;
#pragma unroll
  for (int j = 0; j < 8; j++)
    pk[j] = (unsigned)a[j] | ((unsigned)b[j] << 16);
}

// ---------------------------------------------------------------- casts ----
__global__ void cast_f32_bf16(const float* __restrict__ in,
                              __bf16* __restrict__ out, int n) {
  int i = blockIdx.x * blockDim.x + threadIdx.x;
  if (i < n) out[i] = (__bf16)in[i];
}

// ----------------------------------------------------------------- GEMM ----
// C[M,N] = A[M,K](bf16) * B[K,N](bf16) + bias.  Block tile 128x128, 8 waves,
// each wave computes 32x64 (2 M-subtiles x 4 N-subtiles = 8 WMMAs/k-step).
// MODE 0: write f32 rows to Co.  MODE 1: scatter into q/k/v [B][H][T][64] bf16.
template <int MODE>
__global__ __launch_bounds__(256) void gemm_bf16(
    const __bf16* __restrict__ A, const __bf16* __restrict__ Bm,
    const float* __restrict__ bias, float* __restrict__ Co,
    __bf16* __restrict__ qb, __bf16* __restrict__ kb, __bf16* __restrict__ vb,
    int N, int Kk) {
  constexpr int BM = 128, BN = 128, BK = 32;
  __shared__ __bf16   sA[BM][BK + 8];
  __shared__ unsigned sBp[BK / 2][BN + 4];   // K-pair-packed B tile

  const int tid  = threadIdx.x;
  const int lane = tid & 31, wave = tid >> 5;
  const int lr = lane & 15, hi = lane >> 4;
  const int m0 = (wave & 3) * 32;      // wave's 32-row M strip
  const int n0 = (wave >> 2) * 64;     // wave's 64-wide N strip
  const int blockM = blockIdx.y * BM;
  const int blockN = blockIdx.x * BN;

  v8f acc[2][4] = {};

  const int arow = tid >> 1, acol = (tid & 1) * 16;  // A: 2 thr/row, 16 elem
  const int bp   = tid >> 4, bcol = (tid & 15) * 8;  // B: one k-pair, 8 cols

  for (int k0 = 0; k0 < Kk; k0 += BK) {
    __syncthreads();
    // ---- A tile: async DMA straight into LDS (no VGPR round trip) ----
    const __bf16* ag = &A[(size_t)(blockM + arow) * Kk + k0 + acol];
#if USE_ASYNC_LDS_COPY
    {
      unsigned aoff = lds_offset(&sA[arow][acol]);
      asm volatile("global_load_async_to_lds_b128 %0, %1, off"
                   :: "v"(aoff), "v"((unsigned long long)ag) : "memory");
      asm volatile("global_load_async_to_lds_b128 %0, %1, off"
                   :: "v"(aoff + 16u), "v"((unsigned long long)(ag + 8)) : "memory");
    }
#else
    *(uint4*)&sA[arow][acol]     = *(const uint4*)ag;
    *(uint4*)&sA[arow][acol + 8] = *(const uint4*)(ag + 8);
#endif
    // ---- B tile: load two k-rows, pack bf16 K-pairs, store uint32 ----
    {
      const __bf16* r0 = &Bm[(size_t)(k0 + 2 * bp) * N + blockN + bcol];
      uint4 q0 = *(const uint4*)r0;
      uint4 q1 = *(const uint4*)(r0 + N);
      unsigned pk[8];
      pack8(q0, q1, pk);
      *(uint4*)&sBp[bp][bcol]     = *(uint4*)&pk[0];
      *(uint4*)&sBp[bp][bcol + 4] = *(uint4*)&pk[4];
    }
    if (k0 + BK < Kk) {  // global_prefetch_b8 for the next k-slab
      __builtin_prefetch(&A[(size_t)(blockM + arow) * Kk + k0 + BK + acol], 0, 0);
      __builtin_prefetch(&Bm[(size_t)(k0 + BK + 2 * bp) * N + blockN + bcol], 0, 0);
    }
#if USE_ASYNC_LDS_COPY
    async_wait0();
#endif
    __syncthreads();

    // A fragments (2 M-subtiles): ISA 16-bit A layout, K pairs by lane half
    Frag a[2];
    const int klo = hi * 8;
#pragma unroll
    for (int mi = 0; mi < 2; mi++) {
      const int m = m0 + mi * 16 + lr;
#pragma unroll
      for (int j = 0; j < 4; j++) a[mi].u[j]     = *(const unsigned*)&sA[m][klo + 2 * j];
#pragma unroll
      for (int j = 0; j < 4; j++) a[mi].u[4 + j] = *(const unsigned*)&sA[m][16 + klo + 2 * j];
    }

#pragma unroll
    for (int ns = 0; ns < 4; ns++) {
      Frag b;                       // one B fragment reused by both M-subtiles
      const int n = n0 + ns * 16 + lr;
#pragma unroll
      for (int j = 0; j < 8; j++) b.u[j] = sBp[(j >> 2) * 8 + hi * 4 + (j & 3)][n];
#pragma unroll
      for (int mi = 0; mi < 2; mi++)
        acc[mi][ns] = __builtin_amdgcn_wmma_f32_16x16x32_bf16(
            false, a[mi].v, false, b.v, (short)0, acc[mi][ns], false, false);
    }
  }

  // epilogue: C/D layout — element r maps to M = r + 8*hi, N = lane%16
#pragma unroll
  for (int mi = 0; mi < 2; mi++) {
#pragma unroll
    for (int ns = 0; ns < 4; ns++) {
#pragma unroll
      for (int r = 0; r < 8; r++) {
        const int gm = blockM + m0 + mi * 16 + r + hi * 8;
        const int gn = blockN + n0 + ns * 16 + lr;
        const float val = acc[mi][ns][r] + bias[gn];
        if (MODE == 0) {
          Co[(size_t)gm * N + gn] = val;
        } else {
          const int which = gn >> 10;           // 0=q 1=k 2=v
          const int h = (gn >> 6) & 15;
          const int d = gn & 63;
          const int bidx = gm >> 11;            // / SEQ
          const int t = gm & (SEQ - 1);
          __bf16* dst = (which == 0) ? qb : (which == 1) ? kb : vb;
          dst[(((size_t)bidx * HEADS + h) * SEQ + t) * HD + d] = (__bf16)val;
        }
      }
    }
  }
}

// ------------------------------------------------------ flash attention ----
// One wave32 per (batch, head, 32-row Q tile); streams 32-col KV chunks with
// online softmax (each lane owns one query row). O: 2x4 f32 WMMA fragments.
__global__ __launch_bounds__(32) void flash_attn(
    const __bf16* __restrict__ Q, const __bf16* __restrict__ K,
    const __bf16* __restrict__ V, __bf16* __restrict__ Y) {
  const int lane = threadIdx.x;
  const int lr = lane & 15, hi = lane >> 4;
  const int q0 = blockIdx.x * 32;
  const int h  = blockIdx.y;
  const int b  = blockIdx.z;
  const size_t bh = (size_t)b * HEADS + h;
  const __bf16* Qh = Q + bh * SEQ * HD;
  const __bf16* Kh = K + bh * SEQ * HD;
  const __bf16* Vh = V + bh * SEQ * HD;

  __shared__ float    sS[32][32];
  __shared__ __bf16   sP[32][32];
  __shared__ float    sAlpha[32];
  __shared__ float    sL[32];
  __shared__ unsigned sVp[16][HD + 4];   // K-pair-packed V chunk (32 kv x 64 d)

  // Q fragments: 2 M-subtiles x 2 K-chunks over the 64-dim head
  Frag qf[2][2];
  {
    const int klo = hi * 16;
#pragma unroll
    for (int mi = 0; mi < 2; mi++) {
      const int m = q0 + mi * 16 + lr;
#pragma unroll
      for (int f = 0; f < 2; f++) {
        uint4 c0 = *(const uint4*)&Qh[(size_t)m * HD + f * 32 + klo];
        uint4 c1 = *(const uint4*)&Qh[(size_t)m * HD + f * 32 + 8 + klo];
        qf[mi][f].u[0] = c0.x; qf[mi][f].u[1] = c0.y;
        qf[mi][f].u[2] = c0.z; qf[mi][f].u[3] = c0.w;
        qf[mi][f].u[4] = c1.x; qf[mi][f].u[5] = c1.y;
        qf[mi][f].u[6] = c1.z; qf[mi][f].u[7] = c1.w;
      }
    }
  }

  v8f o[2][4] = {};
  float mRun = -1e30f, lRun = 0.f;  // lane owns query row q0+lane
  const float scale = 0.125f;       // 1/sqrt(64)

  for (int kv0 = 0; kv0 <= q0 + 31; kv0 += 32) {
    __syncthreads();
    // stage V chunk pre-packed: lane = (k-pair p, column half); pack rows
    // 2p and 2p+1 into uint32 K-pairs so PV B-fragments are plain b32 loads.
    {
      const int p = lane & 15, half = lane >> 4, col0 = half * 32;
      const __bf16* r0 = &Vh[(size_t)(kv0 + 2 * p) * HD + col0];
      const __bf16* r1 = r0 + HD;
#pragma unroll
      for (int c = 0; c < 4; c++) {
        uint4 x0 = *(const uint4*)&r0[c * 8];
        uint4 x1 = *(const uint4*)&r1[c * 8];
        unsigned pk[8];
        pack8(x0, x1, pk);
        *(uint4*)&sVp[p][col0 + c * 8]     = *(uint4*)&pk[0];
        *(uint4*)&sVp[p][col0 + c * 8 + 4] = *(uint4*)&pk[4];
      }
    }
    // S = (Q Kᵀ) * scale ; K fragments shared by both M-subtiles
#pragma unroll
    for (int ns = 0; ns < 2; ns++) {
      Frag bf0, bf1;
      const int n = kv0 + ns * 16 + lr;  // kv row for this lane
      const int kb2 = hi * 16;
      {
        uint4 c0 = *(const uint4*)&Kh[(size_t)n * HD + kb2];
        uint4 c1 = *(const uint4*)&Kh[(size_t)n * HD + kb2 + 8];
        bf0.u[0] = c0.x; bf0.u[1] = c0.y; bf0.u[2] = c0.z; bf0.u[3] = c0.w;
        bf0.u[4] = c1.x; bf0.u[5] = c1.y; bf0.u[6] = c1.z; bf0.u[7] = c1.w;
        uint4 c2 = *(const uint4*)&Kh[(size_t)n * HD + 32 + kb2];
        uint4 c3 = *(const uint4*)&Kh[(size_t)n * HD + 32 + kb2 + 8];
        bf1.u[0] = c2.x; bf1.u[1] = c2.y; bf1.u[2] = c2.z; bf1.u[3] = c2.w;
        bf1.u[4] = c3.x; bf1.u[5] = c3.y; bf1.u[6] = c3.z; bf1.u[7] = c3.w;
      }
#pragma unroll
      for (int mi = 0; mi < 2; mi++) {
        v8f s = {};
        s = __builtin_amdgcn_wmma_f32_16x16x32_bf16(
            false, qf[mi][0].v, false, bf0.v, (short)0, s, false, false);
        s = __builtin_amdgcn_wmma_f32_16x16x32_bf16(
            false, qf[mi][1].v, false, bf1.v, (short)0, s, false, false);
#pragma unroll
        for (int r = 0; r < 8; r++)
          sS[mi * 16 + r + hi * 8][ns * 16 + lr] = s[r] * scale;
      }
    }
    __syncthreads();

    // online softmax: every lane owns one query row
    {
      const int r = lane, qrow = q0 + r;
      float sv[32];
      float mc = -1e30f;
#pragma unroll
      for (int c = 0; c < 32; c++) {
        sv[c] = ((kv0 + c) <= qrow) ? sS[r][c] : -1e30f;
        mc = fmaxf(mc, sv[c]);
      }
      const float mNew  = fmaxf(mRun, mc);
      const float alpha = __expf(mRun - mNew);
      float ls = 0.f;
#pragma unroll
      for (int c = 0; c < 32; c++) {
        float p = ((kv0 + c) <= qrow) ? __expf(sv[c] - mNew) : 0.f;
        ls += p;
        sP[r][c] = (__bf16)p;
      }
      lRun = lRun * alpha + ls;
      mRun = mNew;
      sAlpha[r] = alpha;
    }
    __syncthreads();

    // rescale running O by alpha[row]; P fragments per M-subtile
    Frag pf[2];
    const int klo = hi * 8;
#pragma unroll
    for (int mi = 0; mi < 2; mi++) {
      float al[8];
#pragma unroll
      for (int r = 0; r < 8; r++) al[r] = sAlpha[mi * 16 + r + hi * 8];
#pragma unroll
      for (int nd = 0; nd < 4; nd++)
#pragma unroll
        for (int r = 0; r < 8; r++) o[mi][nd][r] *= al[r];
#pragma unroll
      for (int j = 0; j < 4; j++)
        pf[mi].u[j]     = *(const unsigned*)&sP[mi * 16 + lr][klo + 2 * j];
#pragma unroll
      for (int j = 0; j < 4; j++)
        pf[mi].u[4 + j] = *(const unsigned*)&sP[mi * 16 + lr][16 + klo + 2 * j];
    }
    // O += P * V; V fragments (packed b32) shared by both M-subtiles
#pragma unroll
    for (int nd = 0; nd < 4; nd++) {
      Frag bv;
      const int n = nd * 16 + lr;
#pragma unroll
      for (int j = 0; j < 8; j++) bv.u[j] = sVp[(j >> 2) * 8 + hi * 4 + (j & 3)][n];
#pragma unroll
      for (int mi = 0; mi < 2; mi++)
        o[mi][nd] = __builtin_amdgcn_wmma_f32_16x16x32_bf16(
            false, pf[mi].v, false, bv.v, (short)0, o[mi][nd], false, false);
    }
  }

  sL[lane] = lRun;
  __syncthreads();
#pragma unroll
  for (int mi = 0; mi < 2; mi++) {
    float linv[8];
#pragma unroll
    for (int r = 0; r < 8; r++) linv[r] = 1.f / sL[mi * 16 + r + hi * 8];
#pragma unroll
    for (int nd = 0; nd < 4; nd++) {
#pragma unroll
      for (int r = 0; r < 8; r++) {
        const int t = q0 + mi * 16 + r + hi * 8;
        const int dcol = nd * 16 + lr;
        Y[((size_t)b * SEQ + t) * CH + h * HD + dcol] =
            (__bf16)(o[mi][nd][r] * linv[r]);
      }
    }
  }
}

// --------------------------------------------------------------- launch ----
extern "C" void kernel_launch(void* const* d_in, const int* in_sizes, int n_in,
                              void* d_out, int out_size, void* d_ws, size_t ws_size,
                              hipStream_t stream) {
  const float* x      = (const float*)d_in[0];
  const float* w_attn = (const float*)d_in[1];
  const float* b_attn = (const float*)d_in[2];
  const float* w_proj = (const float*)d_in[3];
  const float* b_proj = (const float*)d_in[4];
  float* out = (float*)d_out;

  // workspace carve-up (~92 MB total)
  char* ws = (char*)d_ws;
  __bf16* xb  = (__bf16*)ws; ws += (size_t)MROWS * CH * 2;
  __bf16* wab = (__bf16*)ws; ws += (size_t)CH * NQKV * 2;
  __bf16* wpb = (__bf16*)ws; ws += (size_t)CH * CH * 2;
  __bf16* qb  = (__bf16*)ws; ws += (size_t)MROWS * CH * 2;
  __bf16* kb  = (__bf16*)ws; ws += (size_t)MROWS * CH * 2;
  __bf16* vb  = (__bf16*)ws; ws += (size_t)MROWS * CH * 2;
  __bf16* yb  = (__bf16*)ws; ws += (size_t)MROWS * CH * 2;

  { int n = MROWS * CH;  cast_f32_bf16<<<(n + 255) / 256, 256, 0, stream>>>(x, xb, n); }
  { int n = CH * NQKV;   cast_f32_bf16<<<(n + 255) / 256, 256, 0, stream>>>(w_attn, wab, n); }
  { int n = CH * CH;     cast_f32_bf16<<<(n + 255) / 256, 256, 0, stream>>>(w_proj, wpb, n); }

  // qkv = x @ w_attn + b_attn, scattered to q/k/v [B][H][T][64]
  gemm_bf16<1><<<dim3(NQKV / 128, MROWS / 128), 256, 0, stream>>>(
      xb, wab, b_attn, nullptr, qb, kb, vb, NQKV, CH);

  // causal flash attention -> yb [B][T][C]
  flash_attn<<<dim3(SEQ / 32, HEADS, BATCH), 32, 0, stream>>>(qb, kb, vb, yb);

  // out = y @ w_proj + b_proj (f32)
  gemm_bf16<0><<<dim3(CH / 128, MROWS / 128), 256, 0, stream>>>(
      yb, wpb, b_proj, out, nullptr, nullptr, nullptr, CH, CH);
}
